// MultiHeadAttention_9981503996432
// MI455X (gfx1250) — matrix operand order repaired and checked by reading, not compiled, not run on hardware
//
#include <hip/hip_runtime.h>
#include <hip/hip_bf16.h>
#include <stdint.h>

#define HIDDEN 2048
#define HEADS 16
#define HEAD_DIM 128
#define BATCH 2
#define SEQ 2048

typedef __bf16 v16bf __attribute__((ext_vector_type(16)));
typedef float  v8f   __attribute__((ext_vector_type(8)));

union Frag16 { v16bf v; unsigned u[8]; };

__device__ __forceinline__ unsigned short f32_to_bf16(float f) {
  unsigned u = __float_as_uint(f);
  u += 0x7FFFu + ((u >> 16) & 1u);   // round-to-nearest-even
  return (unsigned short)(u >> 16);
}

// Flat low 32 bits of a generic pointer to a __shared__ object == LDS byte
// address (ISA 10.2).
__device__ __forceinline__ unsigned lds_addr_of(const void* p) {
  return (unsigned)(size_t)p;
}

// GLOBAL_LOAD_ASYNC_TO_LDS_B128: per-lane 16B memory -> LDS, ASYNCcnt-tracked,
// no VGPR data path. Inline asm (portable across ROCm7.2 / clang-23 builtin
// arity differences).
__device__ __forceinline__ void async_load_b128(const void* gptr, void* lptr) {
  asm volatile("global_load_async_to_lds_b128 %0, %1, off"
               :: "v"(lds_addr_of(lptr)),
                  "v"((unsigned long long)(size_t)gptr)
               : "memory");
}
__device__ __forceinline__ void wait_async0() {
  asm volatile("s_wait_asynccnt 0x0" ::: "memory");
}

// ---------------------------------------------------------------------------
// Stage 1: f32 -> bf16 bulk conversion (vectorized float4 -> 4x bf16)
// ---------------------------------------------------------------------------
__global__ __launch_bounds__(256)
void cvt_f32_bf16_kernel(const float* __restrict__ in,
                         unsigned short* __restrict__ out, int n4) {
  int i = blockIdx.x * 256 + threadIdx.x;
  if (i >= n4) return;
  float4 f = reinterpret_cast<const float4*>(in)[i];
  uint2 o;
  o.x = (unsigned)f32_to_bf16(f.x) | ((unsigned)f32_to_bf16(f.y) << 16);
  o.y = (unsigned)f32_to_bf16(f.z) | ((unsigned)f32_to_bf16(f.w) << 16);
  reinterpret_cast<uint2*>(out)[i] = o;
}

// ---------------------------------------------------------------------------
// Stage 2/4: C[M,N] = A[M,K](bf16) * B[N,K]^T(bf16) + bias, WMMA bf16
// Block: 256 threads (8 waves), tile 128x128, K-step 32.
// Wave grid 4(M) x 2(N); each wave computes 32x64 = 2x4 WMMA fragments.
// Double-buffered LDS; tiles staged one step ahead with
// global_load_async_to_lds_b128 (ASYNCcnt) overlapping the WMMAs.
// ---------------------------------------------------------------------------
template<bool OUT_BF16>
__global__ __launch_bounds__(256)
void gemm_nt_bias_kernel(const unsigned short* __restrict__ A,
                         const unsigned short* __restrict__ Bw,
                         const float* __restrict__ bias,
                         void* __restrict__ Cout,
                         int M, int N, int K) {
  __shared__ unsigned short As[2][128][32];
  __shared__ unsigned short Bs[2][128][32];

  const int tid  = threadIdx.x;
  const int lane = tid & 31;
  const int wave = tid >> 5;
  const int hl   = lane >> 4;      // half-wave (0/1)
  const int lr   = lane & 15;
  const int wm   = wave & 3;       // M sub-tile 0..3
  const int wn   = wave >> 2;      // N sub-tile 0..1
  const int bm   = blockIdx.y;
  const int bn   = blockIdx.x;

  const int ldrow = tid >> 1;        // 0..127
  const int ldcol = (tid & 1) * 16;  // 0 / 16 (bf16 elements)

  v8f z = {0.f,0.f,0.f,0.f,0.f,0.f,0.f,0.f};
  v8f acc[2][4];
  #pragma unroll
  for (int mi = 0; mi < 2; ++mi)
    #pragma unroll
    for (int ni = 0; ni < 4; ++ni) acc[mi][ni] = z;

  const unsigned short* arow = A  + (size_t)(bm * 128 + ldrow) * K + ldcol;
  const unsigned short* brow = Bw + (size_t)(bn * 128 + ldrow) * K + ldcol;

  auto issue_tile = [&](int k0, int buf) {
    async_load_b128(arow + k0,     &As[buf][ldrow][ldcol]);
    async_load_b128(arow + k0 + 8, &As[buf][ldrow][ldcol + 8]);
    async_load_b128(brow + k0,     &Bs[buf][ldrow][ldcol]);
    async_load_b128(brow + k0 + 8, &Bs[buf][ldrow][ldcol + 8]);
  };

  issue_tile(0, 0);
  wait_async0();
  __syncthreads();

  for (int k0 = 0; k0 < K; k0 += 32) {
    const int buf = (k0 >> 5) & 1;
    if (k0 + 32 < K) {
      issue_tile(k0 + 32, buf ^ 1);            // overlap with compute below
      __builtin_prefetch(arow + k0 + 64, 0, 3); // global_prefetch_b8, next+1
      __builtin_prefetch(brow + k0 + 64, 0, 3);
    }

    Frag16 af[2], bf[4];
    #pragma unroll
    for (int mi = 0; mi < 2; ++mi) {
      int m = wm * 32 + mi * 16 + lr;
      #pragma unroll
      for (int i = 0; i < 8; ++i) {
        int kk = (i < 4) ? (hl * 8 + 2 * i) : (16 + hl * 8 + 2 * (i - 4));
        af[mi].u[i] = *reinterpret_cast<const unsigned*>(&As[buf][m][kk]);
      }
    }
    #pragma unroll
    for (int ni = 0; ni < 4; ++ni) {
      int n = wn * 64 + ni * 16 + lr;
      #pragma unroll
      for (int i = 0; i < 8; ++i)
        bf[ni].u[i] = *reinterpret_cast<const unsigned*>(&Bs[buf][n][(i < 4) ? (hl * 8 + 2 * i) : (16 + hl * 8 + 2 * (i - 4))]);
    }
    #pragma unroll
    for (int mi = 0; mi < 2; ++mi)
      #pragma unroll
      for (int ni = 0; ni < 4; ++ni)
        acc[mi][ni] = __builtin_amdgcn_wmma_f32_16x16x32_bf16(
            false, af[mi].v, false, bf[ni].v, (short)0, acc[mi][ni], false, false);

    wait_async0();     // next tile resident before anyone proceeds
    __syncthreads();
  }

  #pragma unroll
  for (int mi = 0; mi < 2; ++mi) {
    #pragma unroll
    for (int ni = 0; ni < 4; ++ni) {
      int n = bn * 128 + wn * 64 + ni * 16 + lr;
      float bsv = bias[n];
      #pragma unroll
      for (int j = 0; j < 8; ++j) {
        int m = bm * 128 + wm * 32 + mi * 16 + hl * 8 + j;
        float val = acc[mi][ni][j] + bsv;
        if (OUT_BF16)
          reinterpret_cast<unsigned short*>(Cout)[(size_t)m * N + n] = f32_to_bf16(val);
        else
          reinterpret_cast<float*>(Cout)[(size_t)m * N + n] = val;
      }
    }
  }
}

// ---------------------------------------------------------------------------
// Stage 3: flash attention. Block = (b, h, 128-query slab), 8 waves;
// each wave owns 16 query rows. Online softmax over 32-key tiles.
// Double-buffered tiles: K async-staged to LDS one step ahead; V tile stored
// transposed (VsT[d][key]) so PV B-fragments are contiguous ds_load_b128.
// ---------------------------------------------------------------------------
__global__ __launch_bounds__(256)
void flash_attn_kernel(const unsigned short* __restrict__ Q,
                       const unsigned short* __restrict__ Kmat,
                       const unsigned short* __restrict__ V,
                       unsigned short* __restrict__ Ctx) {
  __shared__ unsigned short Ks[2][32][128];    // [buf][key][d]
  __shared__ unsigned short VsT[2][128][32];   // [buf][d][key] (transposed V)
  __shared__ unsigned short Ps[8][16][32];     // per-wave P staging

  const int tid  = threadIdx.x;
  const int lane = tid & 31;
  const int wave = tid >> 5;
  const int hl   = lane >> 4;
  const int lr   = lane & 15;
  const int qb   = blockIdx.x;
  const int h    = blockIdx.y;
  const int b    = blockIdx.z;
  const size_t base = ((size_t)b * SEQ) * HIDDEN + (size_t)h * HEAD_DIM;
  const int qrow = qb * 128 + wave * 16 + lr;

  const int ldrow = tid >> 3;          // 0..31 keys
  const int ldcol = (tid & 7) * 16;    // 0..112 D (bf16 elems)

  auto issue_k = [&](int kt, int buf) {
    const unsigned short* gk = &Kmat[base + (size_t)(kt * 32 + ldrow) * HIDDEN + ldcol];
    async_load_b128(gk,     &Ks[buf][ldrow][ldcol]);
    async_load_b128(gk + 8, &Ks[buf][ldrow][ldcol + 8]);
  };
  auto stage_v = [&](int kt, int buf) {
    const unsigned short* gv = &V[base + (size_t)(kt * 32 + ldrow) * HIDDEN + ldcol];
    union { uint4 q[2]; unsigned short s[16]; } vv;
    vv.q[0] = reinterpret_cast<const uint4*>(gv)[0];
    vv.q[1] = reinterpret_cast<const uint4*>(gv)[1];
    #pragma unroll
    for (int j = 0; j < 16; ++j)
      VsT[buf][ldcol + j][ldrow] = vv.s[j];
  };

  // Q fragments (16 rows x 128 D) resident in registers (4 A-fragments)
  Frag16 aq[4];
  #pragma unroll
  for (int d = 0; d < 4; ++d)
    #pragma unroll
    for (int i = 0; i < 8; ++i) {
      int kk = d * 32 + ((i < 4) ? (hl * 8 + 2 * i) : (16 + hl * 8 + 2 * (i - 4)));
      aq[d].u[i] = *reinterpret_cast<const unsigned*>(&Q[base + (size_t)qrow * HIDDEN + kk]);
    }

  v8f z = {0.f,0.f,0.f,0.f,0.f,0.f,0.f,0.f};
  float mrow[8], lrow[8];
  v8f o[8];
  #pragma unroll
  for (int j = 0; j < 8; ++j) { mrow[j] = -3.0e38f; lrow[j] = 0.f; }
  #pragma unroll
  for (int nt = 0; nt < 8; ++nt) o[nt] = z;

  const float sscale = 0.12751648f;  // log2(e) / sqrt(128)

  issue_k(0, 0);
  stage_v(0, 0);
  wait_async0();
  __syncthreads();

  for (int kt = 0; kt < SEQ / 32; ++kt) {
    const int buf = kt & 1;
    if (kt + 1 < SEQ / 32) {       // stage next tile, overlapping compute
      issue_k(kt + 1, buf ^ 1);
      stage_v(kt + 1, buf ^ 1);
    }

    // scores: 16 queries x 32 keys = two 16x16 WMMA tiles, K-reduced over D=128
    float sc[2][8];
    #pragma unroll
    for (int sub = 0; sub < 2; ++sub) {
      v8f sf = z;
      #pragma unroll
      for (int d = 0; d < 4; ++d) {
        Frag16 bk;
        int n = sub * 16 + lr;   // key (column) index
        #pragma unroll
        for (int i = 0; i < 8; ++i)
          bk.u[i] = *reinterpret_cast<const unsigned*>(&Ks[buf][n][d * 32 + ((i < 4) ? (hl * 8 + 2 * i) : (16 + hl * 8 + 2 * (i - 4)))]);
        sf = __builtin_amdgcn_wmma_f32_16x16x32_bf16(
            false, aq[d].v, false, bk.v, (short)0, sf, false, false);
      }
      #pragma unroll
      for (int j = 0; j < 8; ++j) sc[sub][j] = sf[j] * sscale;
    }

    // online softmax: rows live as (j + hl*8), columns across 16 lanes of half
    float resc[8];
    #pragma unroll
    for (int j = 0; j < 8; ++j) {
      float tm = fmaxf(sc[0][j], sc[1][j]);
      tm = fmaxf(tm, __shfl_xor(tm, 1));
      tm = fmaxf(tm, __shfl_xor(tm, 2));
      tm = fmaxf(tm, __shfl_xor(tm, 4));
      tm = fmaxf(tm, __shfl_xor(tm, 8));
      float mnew = fmaxf(mrow[j], tm);
      float p0 = exp2f(sc[0][j] - mnew);
      float p1 = exp2f(sc[1][j] - mnew);
      float rs = p0 + p1;
      rs += __shfl_xor(rs, 1);
      rs += __shfl_xor(rs, 2);
      rs += __shfl_xor(rs, 4);
      rs += __shfl_xor(rs, 8);
      float r = exp2f(mrow[j] - mnew);
      lrow[j] = lrow[j] * r + rs;
      mrow[j] = mnew;
      resc[j] = r;
      sc[0][j] = p0;
      sc[1][j] = p1;
    }
    #pragma unroll
    for (int nt = 0; nt < 8; ++nt)
      #pragma unroll
      for (int j = 0; j < 8; ++j) o[nt][j] *= resc[j];

    // C-layout -> A-layout transpose of P through per-wave LDS patch
    #pragma unroll
    for (int sub = 0; sub < 2; ++sub)
      #pragma unroll
      for (int j = 0; j < 8; ++j)
        Ps[wave][hl * 8 + j][sub * 16 + lr] = f32_to_bf16(sc[sub][j]);
    __syncthreads();

    Frag16 pa;
    #pragma unroll
    for (int i = 0; i < 8; ++i) {
      int kk = (i < 4) ? (hl * 8 + 2 * i) : (16 + hl * 8 + 2 * (i - 4));
      pa.u[i] = *reinterpret_cast<const unsigned*>(&Ps[wave][lr][kk]);
    }

    // PV: P(16x32) x V(32x128) -> accumulate into o[0..7]
    // VsT is d-major, so each B-fragment is 2 contiguous ds_load_b128.
    #pragma unroll
    for (int nt = 0; nt < 8; ++nt) {
      Frag16 bv;
      int d = nt * 16 + lr;
      #pragma unroll
      for (int i = 0; i < 8; ++i)
        bv.u[i] = *reinterpret_cast<const unsigned*>(&VsT[buf][d][(i < 4) ? (hl * 8 + 2 * i) : (16 + hl * 8 + 2 * (i - 4))]);
      o[nt] = __builtin_amdgcn_wmma_f32_16x16x32_bf16(
          false, pa.v, false, bv.v, (short)0, o[nt], false, false);
    }

    wait_async0();     // next K tile resident before anyone proceeds
    __syncthreads();
  }

  // finalize: divide by row sums, store context as bf16 in [B,S,HIDDEN] layout
  #pragma unroll
  for (int nt = 0; nt < 8; ++nt) {
    int d = nt * 16 + lr;
    #pragma unroll
    for (int j = 0; j < 8; ++j) {
      int q = qb * 128 + wave * 16 + hl * 8 + j;
      float val = o[nt][j] / lrow[j];
      Ctx[base + (size_t)q * HIDDEN + d] = f32_to_bf16(val);
    }
  }
}

// ---------------------------------------------------------------------------
extern "C" void kernel_launch(void* const* d_in, const int* in_sizes, int n_in,
                              void* d_out, int out_size, void* d_ws, size_t ws_size,
                              hipStream_t stream) {
  (void)in_sizes; (void)n_in; (void)out_size; (void)ws_size;
  const float* X  = (const float*)d_in[0];
  const float* Wq = (const float*)d_in[1];
  const float* bq = (const float*)d_in[2];
  const float* Wk = (const float*)d_in[3];
  const float* bk = (const float*)d_in[4];
  const float* Wv = (const float*)d_in[5];
  const float* bv = (const float*)d_in[6];
  const float* Wo = (const float*)d_in[7];
  const float* bo = (const float*)d_in[8];

  const size_t nX = (size_t)BATCH * SEQ * HIDDEN;  // 8,388,608
  const size_t nW = (size_t)HIDDEN * HIDDEN;       // 4,194,304

  unsigned short* p   = (unsigned short*)d_ws;
  unsigned short* Xb  = p; p += nX;
  unsigned short* Wqb = p; p += nW;
  unsigned short* Wkb = p; p += nW;
  unsigned short* Wvb = p; p += nW;
  unsigned short* Wob = p; p += nW;
  unsigned short* Qb  = p; p += nX;
  unsigned short* Kb  = p; p += nX;
  unsigned short* Vb  = p; p += nX;
  unsigned short* Cb  = p; p += nX;

  // Stage 1: downconvert activations and weights to bf16
  cvt_f32_bf16_kernel<<<(int)(nX / 4 / 256), 256, 0, stream>>>(X,  Xb,  (int)(nX / 4));
  cvt_f32_bf16_kernel<<<(int)(nW / 4 / 256), 256, 0, stream>>>(Wq, Wqb, (int)(nW / 4));
  cvt_f32_bf16_kernel<<<(int)(nW / 4 / 256), 256, 0, stream>>>(Wk, Wkb, (int)(nW / 4));
  cvt_f32_bf16_kernel<<<(int)(nW / 4 / 256), 256, 0, stream>>>(Wv, Wvb, (int)(nW / 4));
  cvt_f32_bf16_kernel<<<(int)(nW / 4 / 256), 256, 0, stream>>>(Wo, Wob, (int)(nW / 4));

  // Stage 2: Q/K/V projections (x @ W.T + b), bf16 outputs
  dim3 gg(HIDDEN / 128, (BATCH * SEQ) / 128);  // (16, 32)
  gemm_nt_bias_kernel<true><<<gg, 256, 0, stream>>>(Xb, Wqb, bq, Qb,
                                                    BATCH * SEQ, HIDDEN, HIDDEN);
  gemm_nt_bias_kernel<true><<<gg, 256, 0, stream>>>(Xb, Wkb, bk, Kb,
                                                    BATCH * SEQ, HIDDEN, HIDDEN);
  gemm_nt_bias_kernel<true><<<gg, 256, 0, stream>>>(Xb, Wvb, bv, Vb,
                                                    BATCH * SEQ, HIDDEN, HIDDEN);

  // Stage 3: flash attention -> context (bf16, [B,S,HIDDEN])
  dim3 ga(SEQ / 128, HEADS, BATCH);            // (16, 16, 2)
  flash_attn_kernel<<<ga, 256, 0, stream>>>(Qb, Kb, Vb, Cb);

  // Stage 4: output projection, f32 result straight to d_out
  gemm_nt_bias_kernel<false><<<gg, 256, 0, stream>>>(Cb, Wob, bo, d_out,
                                                     BATCH * SEQ, HIDDEN, HIDDEN);
}
